// CRU_87754771792016
// MI455X (gfx1250) — matrix operand order repaired and checked by reading, not compiled, not run on hardware
//
#include <hip/hip_runtime.h>
#include <hip/hip_bf16.h>

typedef __bf16 bf16;
typedef __attribute__((ext_vector_type(16))) __bf16 v16bf;
typedef __attribute__((ext_vector_type(8)))  float  v8f;

#define WMMA_BF16(a, b, c) \
  __builtin_amdgcn_wmma_f32_16x16x32_bf16(false, (a), false, (b), (short)0, (c), false, false)

constexpr int Bc  = 2048;
constexpr int Tc  = 512;
constexpr int Dc  = 64;
constexpr int Hc  = 512;
constexpr int NTc = 8;
constexpr int INc = 2 * Dc;        // 128
constexpr int G3H = 3 * Hc;        // 1536
constexpr int BM  = 16;            // batch rows per workgroup
constexpr int NIH = G3H * INc;     // 196608  bf16 elements of W_ih
constexpr int NHH = G3H * Hc;      // 786432  bf16 elements of W_hh
constexpr int NTHR = 512;          // 16 waves per workgroup

// LDS strides (bf16 strides multiple of 4 elements -> 8B-aligned reads)
constexpr int LDH_BF = Hc + 8;     // 520
constexpr int LDI_BF = INc + 8;    // 136
constexpr int LDH_F  = Hc + 1;     // 513

__device__ __forceinline__ float softplusf(float x) {
  float ax = fabsf(x);
  return fmaxf(x, 0.f) + __logf(1.f + __expf(-ax));
}
__device__ __forceinline__ float sigmf(float x) {
  return 1.f / (1.f + __expf(-x));
}

// 16 contiguous bf16 from global (B-matrix fragment: lane-local 32B)
__device__ __forceinline__ v16bf ldg_frag(const bf16* p) {
  union { v16bf v; int4 q[2]; } u;
  u.q[0] = *(const int4*)(p);
  u.q[1] = *(const int4*)(p + 8);
  return u.v;
}

// A-matrix fragment from LDS: contiguous K order,
// lane holds K[c..c+15] (elems 0..15), c = 16*(lane/16)
__device__ __forceinline__ v16bf lds_a_frag(const bf16* p) {
  union { v16bf v; int2 d[4]; } u;
  u.d[0] = *(const int2*)(p);
  u.d[1] = *(const int2*)(p + 4);
  u.d[2] = *(const int2*)(p + 8);
  u.d[3] = *(const int2*)(p + 12);
  return u.v;
}

// ---------------- weight fp32 -> bf16 conversion (one-time) ----------------
__global__ void cru_cvt_weights(const float* __restrict__ Wih,
                                const float* __restrict__ Whh,
                                bf16* __restrict__ out) {
  int i = blockIdx.x * blockDim.x + threadIdx.x;
  if (i < NIH) {
    out[i] = (bf16)Wih[i];
  } else if (i < NIH + NHH) {
    out[i] = (bf16)Whh[i - NIH];
  }
}

// ---------------- persistent CRU recurrence ----------------
__global__ __launch_bounds__(NTHR) void cru_recurrent(
    const float* __restrict__ values,   // [B,T,D]
    const float* __restrict__ mask,     // [B,T,D]
    const float* __restrict__ ts,       // [B,T]
    const float* __restrict__ b_ih,     // [3H]
    const float* __restrict__ b_hh,     // [3H]
    const float* __restrict__ Wdecay,   // [H,1]
    const float* __restrict__ bdecay,   // [H]
    const float* __restrict__ W_head,   // [NT,H]
    const float* __restrict__ b_head,   // [NT]
    const bf16*  __restrict__ wbf0,     // [NIH + NHH] : W_ih then W_hh, bf16
    float* __restrict__ out) {          // [B,NT]

  __shared__ float h_f[BM * LDH_F];     // fp32 master hidden state
  __shared__ bf16  h_b[BM * LDH_BF];    // bf16 copy for WMMA A
  __shared__ bf16  in_b[BM * LDI_BF];   // bf16 concat(v,m) tile

  const int tid  = threadIdx.x;
  const int lane = tid & 31;
  const int wave = tid >> 5;            // 16 waves, 32 h-columns each
  const int colbase = wave * 32;

  // phase-1 mapping: one batch row per wave
  const int  row  = wave;               // 0..15
  const long brow = (long)blockIdx.x * BM + row;

  const int arow  = lane & 15;
  const int ahalf = (lane >> 4) << 3;   // 0 or 8  (C/D row half)
  const int khalf = (lane >> 4) << 4;   // 0 or 16 (B-fragment K half)

  int nIdx[2];
  int offIhR[2], offIhZ[2], offIhN[2];
  int offHhR[2], offHhZ[2], offHhN[2];
  float bR[2], bZ[2], bNi[2], bNh[2];
#pragma unroll
  for (int ct = 0; ct < 2; ++ct) {
    const int j = colbase + ct * 16 + (lane & 15);
    nIdx[ct]   = j;
    offIhR[ct] = j * INc + khalf;
    offIhZ[ct] = (Hc + j) * INc + khalf;
    offIhN[ct] = (2 * Hc + j) * INc + khalf;
    offHhR[ct] = NIH + j * Hc + khalf;
    offHhZ[ct] = NIH + (Hc + j) * Hc + khalf;
    offHhN[ct] = NIH + (2 * Hc + j) * Hc + khalf;
    bR[ct]  = b_ih[j] + b_hh[j];
    bZ[ct]  = b_ih[Hc + j] + b_hh[Hc + j];
    bNi[ct] = b_ih[2 * Hc + j];
    bNh[ct] = b_hh[2 * Hc + j];
  }

  const bf16* aIbase = &in_b[arow * LDI_BF + khalf];
  const bf16* aHbase = &h_b[arow * LDH_BF + khalf];

  // Opaque zero offset, laundered each step: keeps the weight stream inside
  // the t-loop (no LICM hoist -> no spills) WITHOUT destroying the global
  // address-space of the base pointers (so loads lower to global_load_b128,
  // not flat_load_b128 which would contend with the DS pipe / DScnt).
  // The weight image is L2-resident (1.97 MB << 192 MB) and is re-streamed
  // from L2 every step by design.
  int zo = 0;

#pragma unroll 1
  for (int t = 0; t < Tc; ++t) {
    asm volatile("" : "+s"(zo));
    const bf16*  w   = wbf0  + zo;
    const float* wdp = Wdecay + zo;
    const float* bdp = bdecay + zo;

    // ---------- phase 1: decay h, refresh bf16 copy, build input tile ----------
    float dt = 0.f;
    if (t > 0) dt = ts[brow * Tc + t] - ts[brow * Tc + t - 1];

#pragma unroll 4
    for (int cc = 0; cc < 16; ++cc) {
      const int col = lane * 16 + cc;
      float hv;
      if (t == 0) {
        hv = 0.f;                                  // h0 = 0, no decay at first step
      } else {
        const float pre = dt * wdp[col] + bdp[col];
        const float dec = __expf(-softplusf(pre)); // exp(-softplus(.))
        hv = h_f[row * LDH_F + col] * dec;
      }
      h_f[row * LDH_F + col] = hv;
      h_b[row * LDH_BF + col] = (bf16)hv;
    }

    const float* vsrc = values + (brow * Tc + t) * Dc;
    const float* msrc = mask   + (brow * Tc + t) * Dc;
#pragma unroll
    for (int e = 0; e < 4; ++e) {
      const int col = lane * 4 + e;
      const float xv = (col < Dc) ? vsrc[col] : msrc[col - Dc];
      in_b[row * LDI_BF + col] = (bf16)xv;
    }
    if (lane == 0 && t + 1 < Tc) {                 // gfx1250 global_prefetch_b8
      __builtin_prefetch(vsrc + Dc, 0, 0);
      __builtin_prefetch(msrc + Dc, 0, 0);
    }
    __syncthreads();

    // ---------- phase 2: fused gate GEMMs via WMMA ----------
    v8f aR[2] = {}, aZ[2] = {}, aNi[2] = {}, aNh[2] = {};

    // input part: K = 128 (r,z fused input+hidden accumulators; n input part)
#pragma unroll
    for (int kt = 0; kt < INc / 32; ++kt) {
      const v16bf a = lds_a_frag(aIbase + kt * 32);
#pragma unroll
      for (int ct = 0; ct < 2; ++ct) {
        aR[ct]  = WMMA_BF16(a, ldg_frag(w + offIhR[ct] + kt * 32), aR[ct]);
        aZ[ct]  = WMMA_BF16(a, ldg_frag(w + offIhZ[ct] + kt * 32), aZ[ct]);
        aNi[ct] = WMMA_BF16(a, ldg_frag(w + offIhN[ct] + kt * 32), aNi[ct]);
      }
    }

    // hidden part: K = 512
#pragma unroll 4
    for (int kt = 0; kt < Hc / 32; ++kt) {
      const v16bf a = lds_a_frag(aHbase + kt * 32);
#pragma unroll
      for (int ct = 0; ct < 2; ++ct) {
        aR[ct]  = WMMA_BF16(a, ldg_frag(w + offHhR[ct] + kt * 32), aR[ct]);
        aZ[ct]  = WMMA_BF16(a, ldg_frag(w + offHhZ[ct] + kt * 32), aZ[ct]);
        aNh[ct] = WMMA_BF16(a, ldg_frag(w + offHhN[ct] + kt * 32), aNh[ct]);
      }
    }

    // ---------- phase 3: gate nonlinearity + state update ----------
    // C/D layout: lane<16 -> N=lane, M=p ; lane>=16 -> N=lane-16, M=p+8
#pragma unroll
    for (int ct = 0; ct < 2; ++ct) {
      const int j = nIdx[ct];
#pragma unroll
      for (int p = 0; p < 8; ++p) {
        const int m = ahalf + p;
        const float r  = sigmf(aR[ct][p] + bR[ct]);
        const float zg = sigmf(aZ[ct][p] + bZ[ct]);
        const float nn = tanhf(aNi[ct][p] + bNi[ct] + r * (aNh[ct][p] + bNh[ct]));
        const float ho = h_f[m * LDH_F + j];
        h_f[m * LDH_F + j] = (1.f - zg) * nn + zg * ho;
      }
    }
    __syncthreads();
  }

  // ---------- head: out = h_T @ W_head^T + b_head ----------
  if (tid < BM * NTc) {
    const int orow = tid >> 3;
    const int nt   = tid & 7;
    const long b   = (long)blockIdx.x * BM + orow;
    float acc = b_head[nt];
#pragma unroll 8
    for (int k = 0; k < Hc; ++k)
      acc = fmaf(h_f[orow * LDH_F + k], W_head[nt * Hc + k], acc);
    out[b * NTc + nt] = acc;
  }
}

extern "C" void kernel_launch(void* const* d_in, const int* in_sizes, int n_in,
                              void* d_out, int out_size, void* d_ws, size_t ws_size,
                              hipStream_t stream) {
  const float* values  = (const float*)d_in[0];
  const float* mask    = (const float*)d_in[1];
  const float* tstamps = (const float*)d_in[2];
  const float* W_ih    = (const float*)d_in[3];
  const float* W_hh    = (const float*)d_in[4];
  const float* b_ih    = (const float*)d_in[5];
  const float* b_hh    = (const float*)d_in[6];
  const float* W_decay = (const float*)d_in[7];
  const float* b_decay = (const float*)d_in[8];
  const float* W_head  = (const float*)d_in[9];
  const float* b_head  = (const float*)d_in[10];
  float* out = (float*)d_out;
  bf16*  wbf = (bf16*)d_ws;   // NIH + NHH bf16 elements (~1.97 MB)

  const int ncvt = NIH + NHH;
  cru_cvt_weights<<<(ncvt + 255) / 256, 256, 0, stream>>>(W_ih, W_hh, wbf);

  cru_recurrent<<<Bc / BM, NTHR, 0, stream>>>(
      values, mask, tstamps, b_ih, b_hh, W_decay, b_decay,
      W_head, b_head, wbf, out);
}
